// MixtralAttention_2087354105878
// MI455X (gfx1250) — matrix operand order repaired and checked by reading, not compiled, not run on hardware
//
#include <hip/hip_runtime.h>
#include <hip/hip_bf16.h>

#define DEV __device__ __forceinline__

typedef __attribute__((ext_vector_type(16))) __bf16        v16bf;
typedef __attribute__((ext_vector_type(8)))  float         v8f;
typedef __attribute__((ext_vector_type(4)))  unsigned int  u32x4;

constexpr int   T       = 2048;
constexpr int   H       = 32;
constexpr int   KV      = 8;
constexpr int   D       = 128;
constexpr int   HIDDEN  = H * D;              // 4096
constexpr int   QKV_N   = (H + 2 * KV) * D;   // 6144
constexpr int   K_OFF   = H * D;              // 4096
constexpr int   V_OFF   = (H + KV) * D;       // 5120
constexpr float SCALE   = 0.08838834764831845f; // D^-0.5
constexpr int   LDT     = 40;                 // padded stride for 16-row tiles (80B)
constexpr int   LDK     = 136;                // padded stride for 128-wide K rows (272B)

DEV float bf2f(unsigned short u) {
  union { unsigned int i; float f; } v; v.i = ((unsigned int)u) << 16; return v.f;
}
DEV unsigned short f2bf(float f) {
  union { float f; unsigned int i; } v; v.f = f;
  unsigned int r = v.i + 0x7FFFu + ((v.i >> 16) & 1u);
  return (unsigned short)(r >> 16);
}

union FragCvt { u32x4 u[2]; v16bf v; };

// ---------------------------------------------------------------------------
// fp32 -> bf16, 4 elements per thread
// ---------------------------------------------------------------------------
__global__ void cvt_bf16_kernel(const float* __restrict__ src,
                                unsigned short* __restrict__ dst, int n4) {
  int i = blockIdx.x * blockDim.x + threadIdx.x;
  if (i >= n4) return;
  float4 f = ((const float4*)src)[i];
  ushort4 o;
  o.x = f2bf(f.x); o.y = f2bf(f.y); o.z = f2bf(f.z); o.w = f2bf(f.w);
  ((ushort4*)dst)[i] = o;
}

// ---------------------------------------------------------------------------
// C[m,n] = sum_k A[m,k] * W[n,k]   (A: MxK bf16, W: NxK bf16, both row-major)
// Block tile 128x128, BK=32, 8 waves, per-wave 32x64 (2x4 WMMA accumulators).
// Double-buffered LDS pipeline: global->regs for slab kb+1 issued before the
// WMMA block of slab kb, regs->LDS(alt) after, one barrier per iteration.
// ---------------------------------------------------------------------------
template <typename OutT>
__global__ __launch_bounds__(256)
void gemm_nt_kernel(const unsigned short* __restrict__ A,
                    const unsigned short* __restrict__ W,
                    OutT* __restrict__ C, int M, int N, int K) {
  __shared__ __align__(16) unsigned short sA[2][128 * LDT];
  __shared__ __align__(16) unsigned short sW[2][128 * LDT];

  const int tid  = threadIdx.x;
  const int lane = tid & 31, wid = tid >> 5;
  const int l15  = lane & 15, hi  = lane >> 4;
  const int bm   = blockIdx.y * 128, bn = blockIdx.x * 128;
  const int wm   = (wid >> 1) * 32;   // 0,32,64,96
  const int wn   = (wid & 1) * 64;    // 0,64

  // each thread owns 2 x 16B chunks of each tile
  const int r0 = (tid * 2) >> 2,       c0 = ((tid * 2) & 3) * 8;
  const int r1 = (tid * 2 + 1) >> 2,   c1 = ((tid * 2 + 1) & 3) * 8;

  v8f acc[2][4] = {};
  u32x4 ra0, ra1, rw0, rw1;

  const int nK = K / 32;

  // prologue: slab 0 -> regs -> LDS[0]
  ra0 = *(const u32x4*)&A[(size_t)(bm + r0) * K + c0];
  ra1 = *(const u32x4*)&A[(size_t)(bm + r1) * K + c1];
  rw0 = *(const u32x4*)&W[(size_t)(bn + r0) * K + c0];
  rw1 = *(const u32x4*)&W[(size_t)(bn + r1) * K + c1];
  *(u32x4*)&sA[0][r0 * LDT + c0] = ra0;
  *(u32x4*)&sA[0][r1 * LDT + c1] = ra1;
  *(u32x4*)&sW[0][r0 * LDT + c0] = rw0;
  *(u32x4*)&sW[0][r1 * LDT + c1] = rw1;
  __syncthreads();

  for (int kb = 0; kb < nK; ++kb) {
    const int cur = kb & 1, nxt = cur ^ 1;
    const bool more = (kb + 1 < nK);
    if (more) {
      const size_t ko = (size_t)(kb + 1) * 32;
      ra0 = *(const u32x4*)&A[(size_t)(bm + r0) * K + ko + c0];
      ra1 = *(const u32x4*)&A[(size_t)(bm + r1) * K + ko + c1];
      rw0 = *(const u32x4*)&W[(size_t)(bn + r0) * K + ko + c0];
      rw1 = *(const u32x4*)&W[(size_t)(bn + r1) * K + ko + c1];
      __builtin_prefetch(&A[(size_t)(bm + (tid >> 1)) * K + ko + 32], 0, 1);
      __builtin_prefetch(&W[(size_t)(bn + (tid >> 1)) * K + ko + 32], 0, 1);
    }

    v16bf af[2];
    for (int mt = 0; mt < 2; ++mt) {
      FragCvt f;
      const unsigned short* p = &sA[cur][(wm + mt * 16 + l15) * LDT];
      f.u[0] = *(const u32x4*)&p[hi * 8];        // A lane: K = hi*8..+7
      f.u[1] = *(const u32x4*)&p[hi * 8 + 16];   //          K = hi*8+16..+23
      af[mt] = f.v;
    }
    for (int nt = 0; nt < 4; ++nt) {
      FragCvt g;
      const unsigned short* p = &sW[cur][(wn + nt * 16 + l15) * LDT];
      g.u[0] = *(const u32x4*)&p[hi * 8];        // B lane: K = hi*8..+7
      g.u[1] = *(const u32x4*)&p[hi * 8 + 16];   //          K = hi*8+16..+23
      v16bf bf = g.v;
      for (int mt = 0; mt < 2; ++mt)
        acc[mt][nt] = __builtin_amdgcn_wmma_f32_16x16x32_bf16(
            false, af[mt], false, bf, (short)0, acc[mt][nt], false, false);
    }

    if (more) {
      *(u32x4*)&sA[nxt][r0 * LDT + c0] = ra0;
      *(u32x4*)&sA[nxt][r1 * LDT + c1] = ra1;
      *(u32x4*)&sW[nxt][r0 * LDT + c0] = rw0;
      *(u32x4*)&sW[nxt][r1 * LDT + c1] = rw1;
    }
    __syncthreads();
  }

  for (int mt = 0; mt < 2; ++mt)
    for (int nt = 0; nt < 4; ++nt)
      for (int r = 0; r < 8; ++r) {
        int row = bm + wm + mt * 16 + r + 8 * hi;  // C lane row mapping
        int col = bn + wn + nt * 16 + l15;
        float v = acc[mt][nt][r];
        if constexpr (sizeof(OutT) == 2)
          ((unsigned short*)C)[(size_t)row * N + col] = f2bf(v);
        else
          ((float*)C)[(size_t)row * N + col] = v;
      }
}

// ---------------------------------------------------------------------------
// RoPE (in-place on bf16 qkv); folds the softmax scale into Q.
// ---------------------------------------------------------------------------
__global__ void rope_kernel(const int* __restrict__ pos,
                            unsigned short* __restrict__ qkv) {
  int idx = blockIdx.x * blockDim.x + threadIdx.x;
  const int total = T * (H + KV) * 64;
  if (idx >= total) return;
  int i    = idx & 63;
  int head = (idx >> 6) % (H + KV);
  int t    = idx / (64 * (H + KV));

  float p   = (float)pos[t];
  float inv = __expf(-(float)i * (9.210340371976184f / 64.0f)); // 10000^(-i/64)
  float fr  = p * inv;
  float s, c;
  __sincosf(fr, &s, &c);

  bool  isq = head < H;
  int   off = isq ? head * D : K_OFF + (head - H) * D;
  unsigned short* base = qkv + (size_t)t * QKV_N + off;
  float x1 = bf2f(base[i]), x2 = bf2f(base[i + 64]);
  float y1 = x1 * c - x2 * s;
  float y2 = x2 * c + x1 * s;
  if (isq) { y1 *= SCALE; y2 *= SCALE; }
  base[i]      = f2bf(y1);
  base[i + 64] = f2bf(y2);
}

// ---------------------------------------------------------------------------
// Flash attention, causal, GQA (4 q-heads share one kv-head).
// Block = (head, 128 q rows); wave owns 16 q rows; key blocks of 32.
// K and V^T both cooperatively staged in LDS (deduped across the 8 waves);
// all WMMA operands come from registers or ds_read_b128.
// ---------------------------------------------------------------------------
__global__ __launch_bounds__(256)
void attn_kernel(const unsigned short* __restrict__ qkv,
                 unsigned short* __restrict__ out) {
  __shared__ __align__(16) unsigned short sK[32 * LDK];      // K  [key][d]
  __shared__ __align__(16) unsigned short sVT[D * LDT];      // V^T [d][key]
  __shared__ __align__(16) unsigned short sP[8][16 * LDT];   // per-wave P 16x32

  const int tid  = threadIdx.x, lane = tid & 31, wid = tid >> 5;
  const int l15  = lane & 15, hi = lane >> 4;
  const int head = blockIdx.y;
  const int q0   = blockIdx.x * 128 + wid * 16;
  const int kvh  = head / (H / KV);

  // Q fragments (pre-scaled by SCALE in rope): 4 chunks over D=128
  v16bf qf[4];
  {
    const unsigned short* qrow = qkv + (size_t)(q0 + l15) * QKV_N + head * D;
    for (int ch = 0; ch < 4; ++ch) {
      FragCvt f;
      f.u[0] = *(const u32x4*)&qrow[ch * 32 + hi * 8];
      f.u[1] = *(const u32x4*)&qrow[ch * 32 + hi * 8 + 16];
      qf[ch] = f.v;
    }
  }

  v8f  o[8] = {};
  float mrow[8], lrow[8];
  for (int r = 0; r < 8; ++r) { mrow[r] = -1e30f; lrow[r] = 0.0f; }

  const int kbMax = (blockIdx.x + 1) * 4;
  for (int kb = 0; kb < kbMax; ++kb) {
    const int k0 = kb * 32;

    // cooperative stage of K (straight copy) and V (transposed):
    // thread -> key = tid>>3, d0 = (tid&7)*16 (16 elems each)
    {
      int key = tid >> 3;
      int d0  = (tid & 7) * 16;
      const unsigned short* base =
          qkv + (size_t)(k0 + key) * QKV_N + kvh * D + d0;
      u32x4 ka = *(const u32x4*)&base[K_OFF];
      u32x4 kbv = *(const u32x4*)&base[K_OFF + 8];
      u32x4 va = *(const u32x4*)&base[V_OFF];
      u32x4 vb = *(const u32x4*)&base[V_OFF + 8];
      *(u32x4*)&sK[key * LDK + d0]     = ka;
      *(u32x4*)&sK[key * LDK + d0 + 8] = kbv;
      unsigned short tmp[16];
      *(u32x4*)&tmp[0] = va; *(u32x4*)&tmp[8] = vb;
      for (int j = 0; j < 16; ++j) sVT[(d0 + j) * LDT + key] = tmp[j];
    }
    __syncthreads();

    if (k0 <= q0 + 15) {                 // wave has unmasked work this block
      // ---- S = Q K^T : 16 x 32 (B-frags from LDS) ----
      v8f sacc[2] = {};
      for (int nt = 0; nt < 2; ++nt)
        for (int ch = 0; ch < 4; ++ch) {
          const unsigned short* krow = &sK[(nt * 16 + l15) * LDK + ch * 32];
          FragCvt g;
          g.u[0] = *(const u32x4*)&krow[hi * 8];
          g.u[1] = *(const u32x4*)&krow[hi * 8 + 16];
          sacc[nt] = __builtin_amdgcn_wmma_f32_16x16x32_bf16(
              false, qf[ch], false, g.v, (short)0, sacc[nt], false, false);
        }

      // ---- causal mask + online softmax ----
      float p0a[8], p1a[8];
      for (int r = 0; r < 8; ++r) {
        int   qr = q0 + r + 8 * hi;
        float s0 = sacc[0][r];
        float s1 = sacc[1][r];
        if (k0 + l15 > qr)      s0 = -1e30f;
        if (k0 + 16 + l15 > qr) s1 = -1e30f;
        float mx = fmaxf(s0, s1);
        for (int m = 8; m >= 1; m >>= 1) mx = fmaxf(mx, __shfl_xor(mx, m, 32));
        float mnew  = fmaxf(mrow[r], mx);
        float resc  = __expf(mrow[r] - mnew);
        float p0    = __expf(s0 - mnew);
        float p1    = __expf(s1 - mnew);
        float ps    = p0 + p1;
        for (int m = 8; m >= 1; m >>= 1) ps += __shfl_xor(ps, m, 32);
        lrow[r] = lrow[r] * resc + ps;
        mrow[r] = mnew;
        for (int dt = 0; dt < 8; ++dt) o[dt][r] *= resc;
        p0a[r] = p0; p1a[r] = p1;
      }

      // ---- P through per-wave LDS: C-layout -> A-fragment layout ----
      unsigned short* pw = sP[wid];
      for (int r = 0; r < 8; ++r) {
        int m = r + 8 * hi;
        pw[m * LDT + l15]      = f2bf(p0a[r]);
        pw[m * LDT + 16 + l15] = f2bf(p1a[r]);
      }
      FragCvt pf;
      pf.u[0] = *(const u32x4*)&pw[l15 * LDT + hi * 8];
      pf.u[1] = *(const u32x4*)&pw[l15 * LDT + hi * 8 + 16];
      v16bf pa = pf.v;

      // ---- O += P V : 8 d-tiles of 16 ----
      for (int dt = 0; dt < 8; ++dt) {
        const unsigned short* vp = &sVT[(dt * 16 + l15) * LDT];
        FragCvt g;
        g.u[0] = *(const u32x4*)&vp[hi * 8];
        g.u[1] = *(const u32x4*)&vp[hi * 8 + 16];
        o[dt] = __builtin_amdgcn_wmma_f32_16x16x32_bf16(
            false, pa, false, g.v, (short)0, o[dt], false, false);
      }
    }
    __syncthreads();
  }

  // epilogue: normalize and store bf16 attn-out (T x 4096)
  for (int r = 0; r < 8; ++r) {
    float inv = 1.0f / lrow[r];
    int   row = q0 + r + 8 * hi;
    unsigned short* orow = out + (size_t)row * HIDDEN + head * D;
    for (int dt = 0; dt < 8; ++dt)
      orow[dt * 16 + l15] = f2bf(o[dt][r] * inv);
  }
}

// ---------------------------------------------------------------------------
extern "C" void kernel_launch(void* const* d_in, const int* in_sizes, int n_in,
                              void* d_out, int out_size, void* d_ws, size_t ws_size,
                              hipStream_t stream) {
  const int*   positions = (const int*)d_in[0];
  const float* hidden    = (const float*)d_in[1];
  const float* w_qkv     = (const float*)d_in[2];
  const float* w_o       = (const float*)d_in[3];

  unsigned short* hsb   = (unsigned short*)d_ws;                   // 2048x4096
  unsigned short* wqkvb = hsb   + (size_t)T * HIDDEN;              // 6144x4096
  unsigned short* wob   = wqkvb + (size_t)QKV_N * HIDDEN;          // 4096x4096
  unsigned short* qkvb  = wob   + (size_t)HIDDEN * HIDDEN;         // 2048x6144
  unsigned short* attnb = qkvb  + (size_t)T * QKV_N;               // 2048x4096

  int n;
  n = T * HIDDEN / 4;
  cvt_bf16_kernel<<<(n + 255) / 256, 256, 0, stream>>>(hidden, hsb, n);
  n = QKV_N * HIDDEN / 4;
  cvt_bf16_kernel<<<(n + 255) / 256, 256, 0, stream>>>(w_qkv, wqkvb, n);
  n = HIDDEN * HIDDEN / 4;
  cvt_bf16_kernel<<<(n + 255) / 256, 256, 0, stream>>>(w_o, wob, n);

  // qkv = hidden @ w_qkv^T  (NT GEMM, bf16 out)
  gemm_nt_kernel<unsigned short>
      <<<dim3(QKV_N / 128, T / 128), 256, 0, stream>>>(hsb, wqkvb, qkvb, T, QKV_N, HIDDEN);

  n = T * (H + KV) * 64;
  rope_kernel<<<(n + 255) / 256, 256, 0, stream>>>(positions, qkvb);

  attn_kernel<<<dim3(T / 128, H), 256, 0, stream>>>(qkvb, attnb);

  // out = attn @ w_o^T  (w_o stored [HIDDEN][H*D] row-major -> NT GEMM, f32 out)
  gemm_nt_kernel<float>
      <<<dim3(HIDDEN / 128, T / 128), 256, 0, stream>>>(attnb, wob, (float*)d_out, T, HIDDEN, HIDDEN);
}
